// EIGLayer_22874995819130
// MI455X (gfx1250) — matrix operand order repaired and checked by reading, not compiled, not run on hardware
//
#include <hip/hip_runtime.h>
#include <hip/hip_bf16.h>
#include <float.h>
#include <math.h>

#define NN 50000
#define NE 800000
#define FD 64
#define AVGL 2.8332133440562162f   // log(17.0)

typedef __attribute__((ext_vector_type(8)))  float  v8f;
typedef __attribute__((ext_vector_type(16))) __bf16 v16bf;

union U16 { v16bf v; uint4 q[2]; };

static __device__ inline v8f wmma_bf16(v16bf a, v16bf b, v8f c) {
  return __builtin_amdgcn_wmma_f32_16x16x32_bf16(false, a, false, b, (short)0, c, false, false);
}

static __device__ inline void atomicMaxF(float* a, float v) {
  if (v >= 0.f) atomicMax((int*)a, __float_as_int(v));
  else          atomicMin((unsigned int*)a, __float_as_uint(v));
}
static __device__ inline void atomicMinF(float* a, float v) {
  if (v >= 0.f) atomicMin((int*)a, __float_as_int(v));
  else          atomicMax((unsigned int*)a, __float_as_uint(v));
}

// ---------------- workspace layout ----------------
static constexpr size_t AL(size_t x) { return (x + 511) & ~(size_t)511; }
static constexpr size_t NB        = (size_t)NN * FD * 4;       // 12.8 MB
static constexpr size_t OFF_ANODE = 0;
static constexpr size_t OFF_BNODE = OFF_ANODE + NB;
static constexpr size_t OFF_AGG   = OFF_BNODE + NB;            // [N][4][64] f32: sum|sumsq|max|min
static constexpr size_t OFF_DEG   = OFF_AGG   + 4 * NB;
static constexpr size_t OFF_AMP   = AL(OFF_DEG + (size_t)NN * 4);
static constexpr size_t OFF_ATT   = AL(OFF_AMP + (size_t)NN * 4);
static constexpr size_t OFF_CSUM  = AL(OFF_ATT + (size_t)NN * 4);
static constexpr size_t OFF_CSQ   = OFF_CSUM + 256;
static constexpr size_t OFF_AGGBF = AL(OFF_CSQ + 256);                       // N*256 bf16
static constexpr size_t OFF_HBF   = AL(OFF_AGGBF + (size_t)NN * 256 * 2);    // N*64 bf16
static constexpr size_t OFF_YBF   = AL(OFF_HBF   + (size_t)NN * 64 * 2);     // N*64 bf16
static constexpr size_t OFF_PK1   = AL(OFF_YBF   + (size_t)NN * 64 * 2);     // 8192 bf16
static constexpr size_t OFF_PK2   = AL(OFF_PK1 + 8192u * 2);                 // 4096 bf16
static constexpr size_t OFF_PKP1  = AL(OFF_PK2 + 4096u * 2);                 // 53248 bf16
static constexpr size_t OFF_PKP2  = AL(OFF_PKP1 + 53248u * 2);               // 4096 bf16

// ---------------- weight packing (B-operand layout) ----------------
// element i of lane L: k = kt*32 + (L<16?0:8) + i + (i<8?0:8) ; n = nt*16 + (L&15)
__global__ void k_pack(const float* W1, const float* W2, const float* Wp1, const float* Wp2,
                       __bf16* pk1, __bf16* pk2, __bf16* pkp1, __bf16* pkp2) {
  int t = blockIdx.x * 256 + threadIdx.x;
  const int S1 = 2 * 8 * 512, S2 = 2 * 4 * 512, S3 = 26 * 4 * 512, S4 = 2 * 4 * 512;
  if (t < S1) {                                   // W_pre1 as B[64 x 128] (top|bot concat)
    int i = t & 15, lane = (t >> 4) & 31, q = t >> 9;
    int nt = q & 7, kt = q >> 3;
    int k = kt * 32 + ((lane & 16) ? 8 : 0) + i + ((i & 8) ? 8 : 0);
    int n = nt * 16 + (lane & 15);
    float v = (n < 64) ? W1[k * 64 + n] : W1[(64 + k) * 64 + (n - 64)];
    pk1[t] = (__bf16)v;
  } else if (t < S1 + S2) {                       // W_pre2 [64 x 64]
    int tt = t - S1;
    int i = tt & 15, lane = (tt >> 4) & 31, q = tt >> 9;
    int nt = q & 3, kt = q >> 2;
    int k = kt * 32 + ((lane & 16) ? 8 : 0) + i + ((i & 8) ? 8 : 0);
    int n = nt * 16 + (lane & 15);
    pk2[tt] = (__bf16)W2[k * 64 + n];
  } else if (t < S1 + S2 + S3) {                  // W_post1 [832 x 64]
    int tt = t - S1 - S2;
    int i = tt & 15, lane = (tt >> 4) & 31, q = tt >> 9;
    int nt = q & 3, kt = q >> 2;                  // kt 0..25
    int k = kt * 32 + ((lane & 16) ? 8 : 0) + i + ((i & 8) ? 8 : 0);
    int n = nt * 16 + (lane & 15);
    pkp1[tt] = (__bf16)Wp1[k * 64 + n];
  } else if (t < S1 + S2 + S3 + S4) {             // W_post2 [64 x 64]
    int tt = t - S1 - S2 - S3;
    int i = tt & 15, lane = (tt >> 4) & 31, q = tt >> 9;
    int nt = q & 3, kt = q >> 2;
    int k = kt * 32 + ((lane & 16) ? 8 : 0) + i + ((i & 8) ? 8 : 0);
    int n = nt * 16 + (lane & 15);
    pkp2[tt] = (__bf16)Wp2[k * 64 + n];
  }
}

// ---------------- init accumulators ----------------
// agg layout: [N][4][64] = sum | sumsq | max | min
__global__ void k_init(float* agg, float* deg, float* csum, float* csq) {
  int i = blockIdx.x * 256 + threadIdx.x;   // exactly N*256 threads
  int j = i & 255;
  float v = (j < 128) ? 0.f : ((j < 192) ? -FLT_MAX : FLT_MAX);
  agg[i] = v;
  if (i < NN) deg[i] = 0.f;
  if (i < FD) { csum[i] = 0.f; csq[i] = 0.f; }
}

// LDS staging helper: copy cnt uint4's from g to s with all 128 threads.
static __device__ inline void stage(uint4* s, const uint4* g, int cnt) {
  for (int i = threadIdx.x; i < cnt; i += 128) s[i] = g[i];
  __syncthreads();
}

// ---------------- K1: Anode = h@W1_top + b1, Bnode = h@W1_bot ----------------
// 4 tiles per wave, weights staged in LDS (16 KB).
__global__ __launch_bounds__(128) void k_pre(const float* __restrict__ h, const float* __restrict__ b1,
                                             const __bf16* __restrict__ pk1,
                                             float* __restrict__ Anode, float* __restrict__ Bnode) {
  __shared__ __bf16 sW[8192];
  stage((uint4*)sW, (const uint4*)pk1, 8192 * 2 / 16);
  int lane = threadIdx.x & 31;
  int wid = threadIdx.x >> 5;
  int col = lane & 15;
  int ko = (lane & 16) ? 8 : 0;
  int rowOff = (lane & 16) ? 8 : 0;
  for (int t = 0; t < 4; ++t) {
    int tile = (blockIdx.x * 4 + wid) * 4 + t;
    if (tile >= NN / 16) break;                    // wave-uniform
    const float* hr = h + (size_t)(tile * 16 + col) * FD;
    v16bf a[2];
#pragma unroll
    for (int kt = 0; kt < 2; ++kt) {
      int base = kt * 32 + ko;
#pragma unroll
      for (int j = 0; j < 8; ++j) {
        a[kt][j]     = (__bf16)hr[base + j];
        a[kt][j + 8] = (__bf16)hr[base + 16 + j];
      }
    }
    size_t rowBase = (size_t)(tile * 16 + rowOff) * FD;
#pragma unroll
    for (int nt = 0; nt < 8; ++nt) {
      float bias = (nt < 4) ? b1[nt * 16 + col] : 0.f;
      v8f acc;
#pragma unroll
      for (int r = 0; r < 8; ++r) acc[r] = bias;
#pragma unroll
      for (int kt = 0; kt < 2; ++kt) {
        v16bf b = *(const v16bf*)(sW + ((kt * 8 + nt) * 32 + lane) * 16);
        acc = wmma_bf16(a[kt], b, acc);
      }
      float* op = ((nt < 4) ? Anode : Bnode) + rowBase + ((nt < 4) ? nt : nt - 4) * 16 + col;
#pragma unroll
      for (int r = 0; r < 8; ++r) op[r * FD] = acc[r];   // immediate offsets r*256B
    }
  }
}

// ---------------- K2: per-edge e = relu(A[src]+B[dst])@W2 + b2, fused atomic agg ----------------
// 8 edge-tiles per wave, W_pre2 staged in LDS (8 KB).
__global__ __launch_bounds__(128) void k_edge(const int* __restrict__ src, const int* __restrict__ dst,
                                              const float* __restrict__ Anode, const float* __restrict__ Bnode,
                                              const float* __restrict__ b2, const __bf16* __restrict__ pk2,
                                              float* agg, float* deg) {
  __shared__ __bf16 sW[4096];
  stage((uint4*)sW, (const uint4*)pk2, 4096 * 2 / 16);
  int lane = threadIdx.x & 31;
  int wid = threadIdx.x >> 5;
  int col = lane & 15;
  int ko = (lane & 16) ? 8 : 0;
  int off8 = (lane & 16) ? 8 : 0;
  float b2v[4];
#pragma unroll
  for (int nt = 0; nt < 4; ++nt) b2v[nt] = b2[nt * 16 + col];
  for (int t = 0; t < 8; ++t) {
    int tile = (blockIdx.x * 4 + wid) * 8 + t;
    if (tile >= NE / 16) break;                    // wave-uniform
    int e = tile * 16 + col;
    int sn = src[e], dn = dst[e];
    if (lane < 16) atomicAdd(&deg[dn], 1.0f);
    int nodeFor[8];
#pragma unroll
    for (int r = 0; r < 8; ++r) nodeFor[r] = __shfl(dn, r + off8);
    const float* ap = Anode + (size_t)sn * FD;
    const float* bp = Bnode + (size_t)dn * FD;
    v16bf a[2];
#pragma unroll
    for (int kt = 0; kt < 2; ++kt) {
      int base = kt * 32 + ko;
#pragma unroll
      for (int j = 0; j < 8; ++j) {
        float t0 = ap[base + j] + bp[base + j];
        float t1 = ap[base + 16 + j] + bp[base + 16 + j];
        a[kt][j]     = (__bf16)fmaxf(t0, 0.f);
        a[kt][j + 8] = (__bf16)fmaxf(t1, 0.f);
      }
    }
#pragma unroll
    for (int nt = 0; nt < 4; ++nt) {
      v8f acc;
#pragma unroll
      for (int r = 0; r < 8; ++r) acc[r] = 0.f;
#pragma unroll
      for (int kt = 0; kt < 2; ++kt) {
        v16bf b = *(const v16bf*)(sW + ((kt * 4 + nt) * 32 + lane) * 16);
        acc = wmma_bf16(a[kt], b, acc);
      }
#pragma unroll
      for (int r = 0; r < 8; ++r) {
        float ev = acc[r] + b2v[nt];
        // interleaved agg: one base address, four atomics at +0/+64/+128/+192 floats
        float* p = agg + (size_t)nodeFor[r] * 256 + nt * 16 + col;
        atomicAdd(p, ev);
        atomicAdd(p + 64, ev * ev);
        atomicMaxF(p + 128, ev);
        atomicMinF(p + 192, ev);
      }
    }
  }
}

// ---------------- K3: finalize agg (mean/std), bf16 pack, scalers ----------------
__global__ void k_finalize(const float* __restrict__ h, const float* __restrict__ agg,
                           const float* __restrict__ deg,
                           __bf16* aggbf, __bf16* hbf, float* amp, float* att) {
  int i = blockIdx.x * 256 + threadIdx.x;  // exactly N*64
  int n = i >> 6, f = i & 63;
  const float* p = agg + (size_t)n * 256 + f;
  float d = deg[n];
  float inv = 1.0f / d;                    // deg >= 1 guaranteed
  float mean = p[0] * inv;
  float msq  = p[64] * inv;
  float stdv = sqrtf(fmaxf(msq - mean * mean, 0.f) + 1e-5f);
  __bf16* q = aggbf + (size_t)n * 256 + f;
  q[0]   = (__bf16)mean;
  q[64]  = (__bf16)p[128];   // max
  q[128] = (__bf16)p[192];   // min
  q[192] = (__bf16)stdv;
  hbf[i] = (__bf16)h[i];
  if (f == 0) {
    float logD = logf(d + 1.0f);
    amp[n] = logD / AVGL;
    att[n] = AVGL / logD;
  }
}

// ---------------- K4: y = relu(h@Wa + agg@Wb + amp*(agg@Wc) + att*(agg@Wd) + b) ----------------
// W_post1 (104 KB) staged in LDS, 4 tiles per wave.
__global__ __launch_bounds__(128) void k_post1(const __bf16* __restrict__ hbf, const __bf16* __restrict__ aggbf,
                                               const float* __restrict__ amp, const float* __restrict__ att,
                                               const float* __restrict__ bp1, const __bf16* __restrict__ pkp1,
                                               __bf16* __restrict__ ybf) {
  __shared__ __bf16 sW[53248];
  stage((uint4*)sW, (const uint4*)pkp1, 53248 * 2 / 16);
  int lane = threadIdx.x & 31;
  int wid = threadIdx.x >> 5;
  int col = lane & 15;
  int ko = (lane & 16) ? 8 : 0;
  int rowOff = (lane & 16) ? 8 : 0;
  for (int t = 0; t < 4; ++t) {
    int tile = (blockIdx.x * 4 + wid) * 4 + t;
    if (tile >= NN / 16) break;                    // wave-uniform
    int rowA = tile * 16 + col;
    U16 ha[2];
    const __bf16* hr = hbf + (size_t)rowA * 64;
#pragma unroll
    for (int kt = 0; kt < 2; ++kt) {
      int base = kt * 32 + ko;
      ha[kt].q[0] = *(const uint4*)(hr + base);
      ha[kt].q[1] = *(const uint4*)(hr + base + 16);
    }
    U16 ga[8];
    const __bf16* gr = aggbf + (size_t)rowA * 256;
#pragma unroll
    for (int kt = 0; kt < 8; ++kt) {
      int base = kt * 32 + ko;
      ga[kt].q[0] = *(const uint4*)(gr + base);
      ga[kt].q[1] = *(const uint4*)(gr + base + 16);
    }
    float ampv[8], attv[8];
    const float* apz = amp + tile * 16 + rowOff;
    const float* atz = att + tile * 16 + rowOff;
#pragma unroll
    for (int r = 0; r < 8; ++r) { ampv[r] = apz[r]; attv[r] = atz[r]; }
    size_t rowBase = (size_t)(tile * 16 + rowOff) * 64;
#pragma unroll
    for (int nt = 0; nt < 4; ++nt) {
      float bias = bp1[nt * 16 + col];
      v8f acc0, accC, accD;
#pragma unroll
      for (int r = 0; r < 8; ++r) { acc0[r] = bias; accC[r] = 0.f; accD[r] = 0.f; }
#pragma unroll
      for (int kt = 0; kt < 2; ++kt) {   // h @ Wa  (kts 0..1)
        v16bf b = *(const v16bf*)(sW + ((kt * 4 + nt) * 32 + lane) * 16);
        acc0 = wmma_bf16(ha[kt].v, b, acc0);
      }
#pragma unroll
      for (int kt = 0; kt < 8; ++kt) {   // agg @ Wb (kts 2..9)
        v16bf b = *(const v16bf*)(sW + (((2 + kt) * 4 + nt) * 32 + lane) * 16);
        acc0 = wmma_bf16(ga[kt].v, b, acc0);
      }
#pragma unroll
      for (int kt = 0; kt < 8; ++kt) {   // agg @ Wc (kts 10..17)
        v16bf b = *(const v16bf*)(sW + (((10 + kt) * 4 + nt) * 32 + lane) * 16);
        accC = wmma_bf16(ga[kt].v, b, accC);
      }
#pragma unroll
      for (int kt = 0; kt < 8; ++kt) {   // agg @ Wd (kts 18..25)
        v16bf b = *(const v16bf*)(sW + (((18 + kt) * 4 + nt) * 32 + lane) * 16);
        accD = wmma_bf16(ga[kt].v, b, accD);
      }
      __bf16* yp = ybf + rowBase + nt * 16 + col;
#pragma unroll
      for (int r = 0; r < 8; ++r) {
        float y = acc0[r] + ampv[r] * accC[r] + attv[r] * accD[r];
        yp[r * 64] = (__bf16)fmaxf(y, 0.f);        // immediate offsets r*128B
      }
    }
  }
}

// ---------------- K5: out = (y@W_post2 + b)*snorm ; column sums for BN ----------------
__global__ __launch_bounds__(128) void k_post2(const __bf16* __restrict__ ybf, const float* __restrict__ snorm,
                                               const float* __restrict__ bp2, const __bf16* __restrict__ pkp2,
                                               float* __restrict__ out, float* csum, float* csq) {
  __shared__ __bf16 sW[4096];
  stage((uint4*)sW, (const uint4*)pkp2, 4096 * 2 / 16);
  int lane = threadIdx.x & 31;
  int wid = threadIdx.x >> 5;
  int col = lane & 15;
  int ko = (lane & 16) ? 8 : 0;
  int rowOff = (lane & 16) ? 8 : 0;
  for (int t = 0; t < 4; ++t) {
    int tile = (blockIdx.x * 4 + wid) * 4 + t;
    if (tile >= NN / 16) break;                    // wave-uniform
    const __bf16* yr = ybf + (size_t)(tile * 16 + col) * 64;
    U16 a[2];
#pragma unroll
    for (int kt = 0; kt < 2; ++kt) {
      int base = kt * 32 + ko;
      a[kt].q[0] = *(const uint4*)(yr + base);
      a[kt].q[1] = *(const uint4*)(yr + base + 16);
    }
    float snv[8];
    const float* sz = snorm + tile * 16 + rowOff;
#pragma unroll
    for (int r = 0; r < 8; ++r) snv[r] = sz[r];
    size_t rowBase = (size_t)(tile * 16 + rowOff) * 64;
#pragma unroll
    for (int nt = 0; nt < 4; ++nt) {
      float bias = bp2[nt * 16 + col];
      v8f acc;
#pragma unroll
      for (int r = 0; r < 8; ++r) acc[r] = bias;
#pragma unroll
      for (int kt = 0; kt < 2; ++kt) {
        v16bf b = *(const v16bf*)(sW + ((kt * 4 + nt) * 32 + lane) * 16);
        acc = wmma_bf16(a[kt].v, b, acc);
      }
      float* op = out + rowBase + nt * 16 + col;
      float ps = 0.f, pss = 0.f;
#pragma unroll
      for (int r = 0; r < 8; ++r) {
        float z = acc[r] * snv[r];
        op[r * 64] = z;                            // immediate offsets r*256B
        ps += z; pss += z * z;
      }
      atomicAdd(&csum[nt * 16 + col], ps);
      atomicAdd(&csq[nt * 16 + col], pss);
    }
  }
}

// ---------------- K6: batch norm apply (in-place on out) ----------------
__global__ void k_bn(const float* __restrict__ csum, const float* __restrict__ csq,
                     const float* __restrict__ gamma, const float* __restrict__ beta,
                     float* __restrict__ out) {
  int i = blockIdx.x * 256 + threadIdx.x;   // exactly N*64
  int f = i & 63;
  const float invN = 1.0f / (float)NN;
  float mu = csum[f] * invN;
  float var = csq[f] * invN - mu * mu;
  float s = rsqrtf(var + 1e-5f) * gamma[f];
  out[i] = (out[i] - mu) * s + beta[f];
}

extern "C" void kernel_launch(void* const* d_in, const int* in_sizes, int n_in,
                              void* d_out, int out_size, void* d_ws, size_t ws_size,
                              hipStream_t stream) {
  const float* h      = (const float*)d_in[0];
  // d_in[1] = eig (unused by the reference forward)
  const float* snorm  = (const float*)d_in[2];
  const float* W_pre1 = (const float*)d_in[3];
  const float* b_pre1 = (const float*)d_in[4];
  const float* W_pre2 = (const float*)d_in[5];
  const float* b_pre2 = (const float*)d_in[6];
  const float* W_post1= (const float*)d_in[7];
  const float* b_post1= (const float*)d_in[8];
  const float* W_post2= (const float*)d_in[9];
  const float* b_post2= (const float*)d_in[10];
  const float* gamma  = (const float*)d_in[11];
  const float* beta   = (const float*)d_in[12];
  const int*   src    = (const int*)d_in[13];
  const int*   dst    = (const int*)d_in[14];
  float* out = (float*)d_out;

  char* ws = (char*)d_ws;
  float*  Anode = (float*)(ws + OFF_ANODE);
  float*  Bnode = (float*)(ws + OFF_BNODE);
  float*  agg   = (float*)(ws + OFF_AGG);
  float*  deg   = (float*)(ws + OFF_DEG);
  float*  amp   = (float*)(ws + OFF_AMP);
  float*  att   = (float*)(ws + OFF_ATT);
  float*  csum  = (float*)(ws + OFF_CSUM);
  float*  csq   = (float*)(ws + OFF_CSQ);
  __bf16* aggbf = (__bf16*)(ws + OFF_AGGBF);
  __bf16* hbf   = (__bf16*)(ws + OFF_HBF);
  __bf16* ybf   = (__bf16*)(ws + OFF_YBF);
  __bf16* pk1   = (__bf16*)(ws + OFF_PK1);
  __bf16* pk2   = (__bf16*)(ws + OFF_PK2);
  __bf16* pkp1  = (__bf16*)(ws + OFF_PKP1);
  __bf16* pkp2  = (__bf16*)(ws + OFF_PKP2);

  const int NT_N = NN / 16;                   // 3125 node tiles
  const int NT_E = NE / 16;                   // 50000 edge tiles
  const int BLK_N4 = (NT_N + 15) / 16;        // 4 waves * 4 tiles = 16 tiles/block
  const int BLK_E8 = (NT_E + 31) / 32;        // 4 waves * 8 tiles = 32 tiles/block

  k_pack<<<272, 256, 0, stream>>>(W_pre1, W_pre2, W_post1, W_post2, pk1, pk2, pkp1, pkp2);
  k_init<<<NN * 256 / 256, 256, 0, stream>>>(agg, deg, csum, csq);
  k_pre<<<BLK_N4, 128, 0, stream>>>(h, b_pre1, pk1, Anode, Bnode);
  k_edge<<<BLK_E8, 128, 0, stream>>>(src, dst, Anode, Bnode, b_pre2, pk2, agg, deg);
  k_finalize<<<NN * FD / 256, 256, 0, stream>>>(h, agg, deg, aggbf, hbf, amp, att);
  k_post1<<<BLK_N4, 128, 0, stream>>>(hbf, aggbf, amp, att, b_post1, pkp1, ybf);
  k_post2<<<BLK_N4, 128, 0, stream>>>(ybf, snorm, b_post2, pkp2, out, csum, csq);
  k_bn<<<NN * FD / 256, 256, 0, stream>>>(csum, csq, gamma, beta, out);
}
